// TransformerLayer_81166291960254
// MI455X (gfx1250) — compile-verified, matrix operand order checked by reading, not run on hardware
//
#include <hip/hip_runtime.h>
#include <hip/hip_bf16.h>
#include <stdint.h>

// ---------------- problem constants ----------------
#define BB   4
#define SS   1024
#define HHD  1024           // hidden H
#define NHH  16
#define HDD  64
#define II   4096
#define TOK  (BB * SS)      // 4096 tokens
#define LNEPS 1e-5f

// ---------------- WMMA / TDM types ----------------
typedef __attribute__((ext_vector_type(16))) __bf16 v16bf;
typedef __attribute__((ext_vector_type(8)))  __bf16 v8bf;
typedef __attribute__((ext_vector_type(4)))  __bf16 v4bf;
typedef __attribute__((ext_vector_type(8)))  float  v8f;
typedef __attribute__((ext_vector_type(4)))  unsigned v4u;
typedef __attribute__((ext_vector_type(8)))  int    v8i;
typedef __attribute__((ext_vector_type(4)))  int    v4i;

union Frag { v16bf v; unsigned u[8]; uint4 q[2]; };
union BfBits { __bf16 b; unsigned short u; };

__device__ __forceinline__ v8f v8f_zero() {
  v8f z;
#pragma unroll
  for (int i = 0; i < 8; ++i) z[i] = 0.0f;
  return z;
}

// Load a 16x32 bf16 A/B fragment from LDS (row-major, rsU32 = row stride in dwords).
// Lane l holds row (passed in), K pattern: u[v] = base + (v<4?0:8) + (lane>=16)*4 + (v&3).
__device__ __forceinline__ void load_frag(const unsigned* lds, int rsU32, int row,
                                          int kkU32, int lane, Frag& f) {
  const unsigned* p = lds + row * rsU32 + kkU32 + ((lane >> 4) << 2);
  f.q[0] = *(const uint4*)p;
  f.q[1] = *(const uint4*)(p + 8);
}

__device__ __forceinline__ v8f wmma_bf16(const Frag& a, const Frag& b, v8f c) {
  return __builtin_amdgcn_wmma_f32_16x16x32_bf16(false, a.v, false, b.v,
                                                 (short)0, c, false, false);
}

// ---------------- TDM: DMA a 128-row x 32-col bf16 tile (row stride K elems) into LDS.
// D# per CDNA5 ISA sec.8. LDS pad: after every 16 DW (=64B row) add 4 DW
// -> LDS row stride 80B = 40 bf16 = 20 dwords, matching load_frag's swizzle.
__device__ __forceinline__ void tdm_load_tile_128x32(unsigned lds_addr,
                                                     const __bf16* gptr, int K) {
  uint64_t ga = (uint64_t)(uintptr_t)gptr;
  unsigned uk = (unsigned)K;
  v4u g0;
  g0[0] = 1u;                                              // count=1, user desc
  g0[1] = lds_addr;                                        // LDS byte address
  g0[2] = (unsigned)ga;                                    // global_addr[31:0]
  g0[3] = (unsigned)((ga >> 32) & 0x01FFFFFFu) | (2u << 30); // addr[56:32] | type=2
  v8i g1;
  g1[0] = (int)(0x00010000u          // data_size=1 (2 bytes)
              | (1u << 20)           // pad_enable
              | (3u << 22)           // pad_interval: 16 DWORDs
              | (3u << 25));         // pad_amount:   4 DWORDs
  g1[1] = (int)((uk & 0xFFFFu) << 16);          // tensor_dim0 lo16 @ bits63:48
  g1[2] = (int)((uk >> 16) | (128u << 16));     // tensor_dim0 hi16 | tensor_dim1 lo16
  g1[3] = (int)(32u << 16);                     // tensor_dim1 hi16=0 | tile_dim0=32
  g1[4] = (int)128u;                            // tile_dim1=128, tile_dim2=0
  g1[5] = (int)uk;                              // tensor_dim0_stride lo32
  g1[6] = 0;                                    // stride hi16 | dim1_stride lo16
  g1[7] = 0;
  v4i z4 = {0, 0, 0, 0};
#if defined(__clang_major__) && (__clang_major__ >= 23)
  v8i z8 = {0, 0, 0, 0, 0, 0, 0, 0};
  __builtin_amdgcn_tensor_load_to_lds(g0, g1, z4, z4, z8, 0);
#else
  __builtin_amdgcn_tensor_load_to_lds(g0, g1, z4, z4, 0);
#endif
}

// ---------------- weight transpose + f32->bf16 convert: out[n][k] = in[k][n] ----------------
__global__ void k_transpose_bf16(const float* __restrict__ in, __bf16* __restrict__ out,
                                 int R, int C) {
  __shared__ float t[32][33];
  int c0 = blockIdx.x * 32, r0 = blockIdx.y * 32;
#pragma unroll
  for (int i = threadIdx.y; i < 32; i += 8)
    t[i][threadIdx.x] = in[(size_t)(r0 + i) * C + (c0 + threadIdx.x)];
  __syncthreads();
#pragma unroll
  for (int i = threadIdx.y; i < 32; i += 8)
    out[(size_t)(c0 + i) * R + (r0 + threadIdx.x)] = (__bf16)t[threadIdx.x][i];
}

// ---------------- LayerNorm (H=1024) -> bf16, one row per 256-thread block ----------------
__global__ void k_layernorm_bf16(const float* __restrict__ x, const float* __restrict__ w,
                                 __bf16* __restrict__ out) {
  int row = blockIdx.x, tid = threadIdx.x, lane = tid & 31, wvi = tid >> 5;
  size_t base = (size_t)row * HHD;
  float4 xv = ((const float4*)(x + base))[tid];
  float s  = xv.x + xv.y + xv.z + xv.w;
  float ss = xv.x * xv.x + xv.y * xv.y + xv.z * xv.z + xv.w * xv.w;
#pragma unroll
  for (int o = 16; o; o >>= 1) { s += __shfl_xor(s, o); ss += __shfl_xor(ss, o); }
  __shared__ float r0[8], r1[8];
  if (lane == 0) { r0[wvi] = s; r1[wvi] = ss; }
  __syncthreads();
  float ts = 0.f, tss = 0.f;
#pragma unroll
  for (int i = 0; i < 8; ++i) { ts += r0[i]; tss += r1[i]; }
  float mu   = ts * (1.0f / HHD);
  float var  = tss * (1.0f / HHD) - mu * mu;
  float rstd = rsqrtf(var + LNEPS);
  float4 wv4 = ((const float4*)w)[tid];
  v4bf o;
  o[0] = (__bf16)((xv.x - mu) * rstd * wv4.x);
  o[1] = (__bf16)((xv.y - mu) * rstd * wv4.y);
  o[2] = (__bf16)((xv.z - mu) * rstd * wv4.z);
  o[3] = (__bf16)((xv.w - mu) * rstd * wv4.w);
  *(v4bf*)(out + base + tid * 4) = o;
}

// ---------------- bf16 WMMA GEMM with TDM tile DMA: C[M,N] = A[M,K] * Bt[N,K]^T -------------
// 256 threads = 8 waves; block tile 128x128, K-step 32, double-buffered LDS.
// Wave 0 issues tensor_load_to_lds for the next K tile while all waves run WMMA on the
// current one; s_wait_tensorcnt + workgroup barriers synchronize producer/consumer.
template <bool OUT_BF16>
__global__ void __launch_bounds__(256)
k_gemm_bf16(const __bf16* __restrict__ A, const __bf16* __restrict__ Bt,
            void* __restrict__ Cout, const float* __restrict__ res,
            float scale, int M, int N, int K) {
  __shared__ unsigned lds_a[2][128 * 20];   // 128 rows x 32 bf16 + 8 bf16 TDM pad = 20 dw
  __shared__ unsigned lds_b[2][128 * 20];
  int tid = threadIdx.x, lane = tid & 31, wave = tid >> 5;
  int l16 = lane & 15, hf = lane >> 4;
  int wm = wave >> 2, wn = wave & 3;
  int mBase = blockIdx.y * 128, nBase = blockIdx.x * 128;
  int nk = K >> 5;

  const __bf16* aTile = A  + (size_t)mBase * K;
  const __bf16* bTile = Bt + (size_t)nBase * K;

  v8f acc[4][2];
#pragma unroll
  for (int mt = 0; mt < 4; ++mt)
#pragma unroll
    for (int nt = 0; nt < 2; ++nt) acc[mt][nt] = v8f_zero();

  if (wave == 0) {  // prologue: DMA first tiles, drain TENSORcnt
    tdm_load_tile_128x32((unsigned)(uintptr_t)&lds_a[0][0], aTile, K);
    tdm_load_tile_128x32((unsigned)(uintptr_t)&lds_b[0][0], bTile, K);
    __builtin_amdgcn_s_wait_tensorcnt(0);
  }
  __syncthreads();
  asm volatile("" ::: "memory");   // LDS produced by TDM, not by this shader

  for (int kb = 0; kb < nk; ++kb) {
    int p = kb & 1;
    if (wave == 0 && kb + 1 < nk) {  // overlap next-tile DMA with WMMA on current tile
      tdm_load_tile_128x32((unsigned)(uintptr_t)&lds_a[p ^ 1][0], aTile + (kb + 1) * 32, K);
      tdm_load_tile_128x32((unsigned)(uintptr_t)&lds_b[p ^ 1][0], bTile + (kb + 1) * 32, K);
    }
    Frag af[4];
#pragma unroll
    for (int mt = 0; mt < 4; ++mt)
      load_frag(lds_a[p], 20, wm * 64 + mt * 16 + l16, 0, lane, af[mt]);
#pragma unroll
    for (int nt = 0; nt < 2; ++nt) {
      Frag bfr;
      load_frag(lds_b[p], 20, wn * 32 + nt * 16 + l16, 0, lane, bfr);
#pragma unroll
      for (int mt = 0; mt < 4; ++mt) acc[mt][nt] = wmma_bf16(af[mt], bfr, acc[mt][nt]);
    }
    __syncthreads();                 // all waves done reading buffer p
    if (wave == 0 && kb + 1 < nk) __builtin_amdgcn_s_wait_tensorcnt(0);
    __syncthreads();                 // buffer p^1 now valid for everyone
    asm volatile("" ::: "memory");
  }

#pragma unroll
  for (int mt = 0; mt < 4; ++mt)
#pragma unroll
    for (int nt = 0; nt < 2; ++nt)
#pragma unroll
      for (int r = 0; r < 8; ++r) {
        int m = mBase + wm * 64 + mt * 16 + hf * 8 + r;
        int n = nBase + wn * 32 + nt * 16 + l16;
        size_t idx = (size_t)m * N + n;
        float v = acc[mt][nt][r];
        if constexpr (OUT_BF16) {
          ((__bf16*)Cout)[idx] = (__bf16)(v * scale);
        } else {
          float o = v;
          if (res) o += res[idx];
          ((float*)Cout)[idx] = o;
        }
      }
}

// ---------------- flash attention: one (b,h, 64-row q tile) per 128-thread block ----------------
__global__ void __launch_bounds__(128)
k_attention(const __bf16* __restrict__ Q, const __bf16* __restrict__ K,
            const __bf16* __restrict__ V, __bf16* __restrict__ O) {
  __shared__ unsigned q_lds[64 * 36];    // 64 x 64 bf16, stride 72 bf16 = 36 dw
  __shared__ unsigned k_lds[64 * 36];
  __shared__ unsigned vt_lds[64 * 36];   // transposed: row = d, col = key
  __shared__ unsigned p_lds[64 * 36];    // per-wave 16-row P scratch
  int bh = blockIdx.y, b = bh >> 4, h = bh & 15;
  int qt = blockIdx.x;
  int tid = threadIdx.x, lane = tid & 31, wave = tid >> 5;
  int l16 = lane & 15, hf = lane >> 4;
  int colbase = h * HDD;
  int row2 = tid >> 1, hc = tid & 1;

  { // load Q tile (Q already scaled by 1/sqrt(HD) at projection)
    const uint4* gp = (const uint4*)(Q + (size_t)(b * SS + qt * 64 + row2) * HHD + colbase + hc * 32);
    uint4 x0 = gp[0], x1 = gp[1], x2 = gp[2], x3 = gp[3];
    uint4* lp = (uint4*)&q_lds[row2 * 36 + hc * 16];
    lp[0] = x0; lp[1] = x1; lp[2] = x2; lp[3] = x3;
  }
  __syncthreads();
  Frag aq0, aq1;
  load_frag(q_lds, 36, wave * 16 + l16, 0, lane, aq0);
  load_frag(q_lds, 36, wave * 16 + l16, 16, lane, aq1);

  v8f o[4];
  float m_run[8], l_run[8];
#pragma unroll
  for (int dt = 0; dt < 4; ++dt) o[dt] = v8f_zero();
#pragma unroll
  for (int r = 0; r < 8; ++r) { m_run[r] = -1e30f; l_run[r] = 0.0f; }

  for (int kt = 0; kt < SS / 64; ++kt) {
    __syncthreads();  // previous iteration's WMMA reads of k/vt complete
    { // K tile
      const uint4* gp = (const uint4*)(K + (size_t)(b * SS + kt * 64 + row2) * HHD + colbase + hc * 32);
      uint4 x0 = gp[0], x1 = gp[1], x2 = gp[2], x3 = gp[3];
      uint4* lp = (uint4*)&k_lds[row2 * 36 + hc * 16];
      lp[0] = x0; lp[1] = x1; lp[2] = x2; lp[3] = x3;
    }
    { // V tile, stored transposed (d-major) for PV B-fragments
      const uint4* gp = (const uint4*)(V + (size_t)(b * SS + kt * 64 + row2) * HHD + colbase + hc * 32);
      uint4 x0 = gp[0], x1 = gp[1], x2 = gp[2], x3 = gp[3];
      unsigned vals[16] = { x0.x, x0.y, x0.z, x0.w, x1.x, x1.y, x1.z, x1.w,
                            x2.x, x2.y, x2.z, x2.w, x3.x, x3.y, x3.z, x3.w };
      unsigned short* vt16 = (unsigned short*)vt_lds;
#pragma unroll
      for (int j = 0; j < 16; ++j) {
        int d = hc * 32 + j * 2;
        vt16[d * 72 + row2]       = (unsigned short)(vals[j] & 0xffffu);
        vt16[(d + 1) * 72 + row2] = (unsigned short)(vals[j] >> 16);
      }
    }
    if (kt + 1 < SS / 64) {   // global_prefetch_b8 on next tiles
      __builtin_prefetch(K + (size_t)(b * SS + (kt + 1) * 64 + row2) * HHD + colbase + hc * 32, 0, 1);
      __builtin_prefetch(V + (size_t)(b * SS + (kt + 1) * 64 + row2) * HHD + colbase + hc * 32, 0, 1);
    }
    __syncthreads();

    // scores S = Q Kt : 16 q-rows x 64 keys per wave (4 WMMA n-tiles, 2 K-steps of HD)
    v8f s[4];
#pragma unroll
    for (int nt = 0; nt < 4; ++nt) {
      s[nt] = v8f_zero();
      Frag bk;
      load_frag(k_lds, 36, nt * 16 + l16, 0, lane, bk);
      s[nt] = wmma_bf16(aq0, bk, s[nt]);
      load_frag(k_lds, 36, nt * 16 + l16, 16, lane, bk);
      s[nt] = wmma_bf16(aq1, bk, s[nt]);
    }

    // streaming softmax over 64 keys; row r lives in the 16-lane half (hf)
    unsigned short* p16 = (unsigned short*)p_lds;
#pragma unroll
    for (int r = 0; r < 8; ++r) {
      float tmax = fmaxf(fmaxf(s[0][r], s[1][r]), fmaxf(s[2][r], s[3][r]));
#pragma unroll
      for (int ofs = 1; ofs < 16; ofs <<= 1) tmax = fmaxf(tmax, __shfl_xor(tmax, ofs));
      float mnew  = fmaxf(m_run[r], tmax);
      float alpha = __expf(m_run[r] - mnew);
      m_run[r] = mnew;
      float lsum = 0.0f;
#pragma unroll
      for (int nt = 0; nt < 4; ++nt) {
        float p = __expf(s[nt][r] - mnew);
        s[nt][r] = p; lsum += p;
      }
#pragma unroll
      for (int ofs = 1; ofs < 16; ofs <<= 1) lsum += __shfl_xor(lsum, ofs);
      l_run[r] = l_run[r] * alpha + lsum;
#pragma unroll
      for (int dt = 0; dt < 4; ++dt) o[dt][r] *= alpha;
      int prow = wave * 16 + hf * 8 + r;
#pragma unroll
      for (int nt = 0; nt < 4; ++nt) {
        BfBits pb; pb.b = (__bf16)s[nt][r];
        p16[prow * 72 + nt * 16 + l16] = pb.u;
      }
    }

    // O += P V : P is wave-local in LDS (same-wave DS ordering), V transposed in LDS
    Frag ap0, ap1;
    load_frag(p_lds, 36, wave * 16 + l16, 0, lane, ap0);
    load_frag(p_lds, 36, wave * 16 + l16, 16, lane, ap1);
#pragma unroll
    for (int dt = 0; dt < 4; ++dt) {
      Frag bv;
      load_frag(vt_lds, 36, dt * 16 + l16, 0, lane, bv);
      o[dt] = wmma_bf16(ap0, bv, o[dt]);
      load_frag(vt_lds, 36, dt * 16 + l16, 16, lane, bv);
      o[dt] = wmma_bf16(ap1, bv, o[dt]);
    }
  }

#pragma unroll
  for (int r = 0; r < 8; ++r) {
    float inv = 1.0f / l_run[r];
    size_t token = (size_t)(b * SS + qt * 64 + wave * 16 + hf * 8 + r);
#pragma unroll
    for (int dt = 0; dt < 4; ++dt)
      O[token * HHD + colbase + dt * 16 + l16] = (__bf16)(o[dt][r] * inv);
  }
}

// ---------------- GeGLU (exact gelu) + mid LayerNorm over I=4096 -> bf16 ----------------
__global__ void k_geglu_midln(const __bf16* __restrict__ U, const __bf16* __restrict__ Vg,
                              const float* __restrict__ w, __bf16* __restrict__ out) {
  int row = blockIdx.x, tid = threadIdx.x, lane = tid & 31, wvi = tid >> 5;
  size_t base = (size_t)row * II;
  v8bf u0 = ((const v8bf*)(U + base))[tid * 2];
  v8bf u1 = ((const v8bf*)(U + base))[tid * 2 + 1];
  v8bf g0 = ((const v8bf*)(Vg + base))[tid * 2];
  v8bf g1 = ((const v8bf*)(Vg + base))[tid * 2 + 1];
  float g[16];
  float s = 0.f, ss = 0.f;
#pragma unroll
  for (int j = 0; j < 8; ++j) {
    float uu = (float)u0[j], vv = (float)g0[j];
    float gg = 0.5f * uu * (1.0f + erff(uu * 0.70710678118f)) * vv;
    g[j] = gg; s += gg; ss += gg * gg;
  }
#pragma unroll
  for (int j = 0; j < 8; ++j) {
    float uu = (float)u1[j], vv = (float)g1[j];
    float gg = 0.5f * uu * (1.0f + erff(uu * 0.70710678118f)) * vv;
    g[8 + j] = gg; s += gg; ss += gg * gg;
  }
#pragma unroll
  for (int o = 16; o; o >>= 1) { s += __shfl_xor(s, o); ss += __shfl_xor(ss, o); }
  __shared__ float r0[8], r1[8];
  if (lane == 0) { r0[wvi] = s; r1[wvi] = ss; }
  __syncthreads();
  float ts = 0.f, tss = 0.f;
#pragma unroll
  for (int i = 0; i < 8; ++i) { ts += r0[i]; tss += r1[i]; }
  float mu   = ts * (1.0f / II);
  float var  = tss * (1.0f / II) - mu * mu;
  float rstd = rsqrtf(var + LNEPS);
  int col0 = tid * 16;
  v8bf o0, o1;
#pragma unroll
  for (int j = 0; j < 8; ++j) o0[j] = (__bf16)((g[j] - mu) * rstd * w[col0 + j]);
#pragma unroll
  for (int j = 0; j < 8; ++j) o1[j] = (__bf16)((g[8 + j] - mu) * rstd * w[col0 + 8 + j]);
  ((v8bf*)(out + base))[tid * 2]     = o0;
  ((v8bf*)(out + base))[tid * 2 + 1] = o1;
}

// ---------------- host-side launcher ----------------
static inline char* ws_take(char*& p, size_t bytes) {
  char* r = (char*)(((uintptr_t)p + 255u) & ~(uintptr_t)255u);
  p = r + bytes;
  return r;
}

extern "C" void kernel_launch(void* const* d_in, const int* in_sizes, int n_in,
                              void* d_out, int out_size, void* d_ws, size_t ws_size,
                              hipStream_t stream) {
  (void)in_sizes; (void)n_in; (void)out_size; (void)ws_size;
  const float* x       = (const float*)d_in[0];
  const float* ln_attn = (const float*)d_in[1];
  const float* wq      = (const float*)d_in[2];
  const float* wk      = (const float*)d_in[3];
  const float* wv      = (const float*)d_in[4];
  const float* wo      = (const float*)d_in[5];
  const float* ln_pre  = (const float*)d_in[6];
  const float* wi0     = (const float*)d_in[7];
  const float* wi1     = (const float*)d_in[8];
  const float* ln_mid  = (const float*)d_in[9];
  const float* wout    = (const float*)d_in[10];

  char* wp = (char*)d_ws;
  __bf16* wqT   = (__bf16*)ws_take(wp, (size_t)HHD * HHD * 2);
  __bf16* wkT   = (__bf16*)ws_take(wp, (size_t)HHD * HHD * 2);
  __bf16* wvT   = (__bf16*)ws_take(wp, (size_t)HHD * HHD * 2);
  __bf16* woT   = (__bf16*)ws_take(wp, (size_t)HHD * HHD * 2);
  __bf16* wi0T  = (__bf16*)ws_take(wp, (size_t)II * HHD * 2);
  __bf16* wi1T  = (__bf16*)ws_take(wp, (size_t)II * HHD * 2);
  __bf16* woutT = (__bf16*)ws_take(wp, (size_t)HHD * II * 2);
  __bf16* h1    = (__bf16*)ws_take(wp, (size_t)TOK * HHD * 2);
  __bf16* qb    = (__bf16*)ws_take(wp, (size_t)TOK * HHD * 2);
  __bf16* kb    = (__bf16*)ws_take(wp, (size_t)TOK * HHD * 2);
  __bf16* vb    = (__bf16*)ws_take(wp, (size_t)TOK * HHD * 2);
  __bf16* aob   = (__bf16*)ws_take(wp, (size_t)TOK * HHD * 2);
  float*  x1    = (float*) ws_take(wp, (size_t)TOK * HHD * 4);
  __bf16* h2    = (__bf16*)ws_take(wp, (size_t)TOK * HHD * 2);
  __bf16* ub    = (__bf16*)ws_take(wp, (size_t)TOK * II * 2);
  __bf16* vgb   = (__bf16*)ws_take(wp, (size_t)TOK * II * 2);
  __bf16* gb    = (__bf16*)ws_take(wp, (size_t)TOK * II * 2);

  dim3 tb(32, 8);
  // weight transposes + bf16 convert: out[n][k] = in[k][n]
  k_transpose_bf16<<<dim3(HHD / 32, HHD / 32), tb, 0, stream>>>(wq,   wqT,   HHD, HHD);
  k_transpose_bf16<<<dim3(HHD / 32, HHD / 32), tb, 0, stream>>>(wk,   wkT,   HHD, HHD);
  k_transpose_bf16<<<dim3(HHD / 32, HHD / 32), tb, 0, stream>>>(wv,   wvT,   HHD, HHD);
  k_transpose_bf16<<<dim3(HHD / 32, HHD / 32), tb, 0, stream>>>(wo,   woT,   HHD, HHD);
  k_transpose_bf16<<<dim3(II / 32,  HHD / 32), tb, 0, stream>>>(wi0,  wi0T,  HHD, II);
  k_transpose_bf16<<<dim3(II / 32,  HHD / 32), tb, 0, stream>>>(wi1,  wi1T,  HHD, II);
  k_transpose_bf16<<<dim3(HHD / 32, II / 32),  tb, 0, stream>>>(wout, woutT, II,  HHD);

  // pre-attention LN
  k_layernorm_bf16<<<TOK, 256, 0, stream>>>(x, ln_attn, h1);

  // QKV projections (Q folded with 1/sqrt(HD) = 0.125 before bf16 quantization)
  dim3 gH(HHD / 128, TOK / 128);
  k_gemm_bf16<true><<<gH, 256, 0, stream>>>(h1, wqT, qb, nullptr, 0.125f, TOK, HHD, HHD);
  k_gemm_bf16<true><<<gH, 256, 0, stream>>>(h1, wkT, kb, nullptr, 1.0f,   TOK, HHD, HHD);
  k_gemm_bf16<true><<<gH, 256, 0, stream>>>(h1, wvT, vb, nullptr, 1.0f,   TOK, HHD, HHD);

  // flash attention
  k_attention<<<dim3(SS / 64, BB * NHH), 128, 0, stream>>>(qb, kb, vb, aob);

  // output projection + residual
  k_gemm_bf16<false><<<gH, 256, 0, stream>>>(aob, woT, x1, x, 1.0f, TOK, HHD, HHD);

  // pre-MLP LN
  k_layernorm_bf16<<<TOK, 256, 0, stream>>>(x1, ln_pre, h2);

  // MLP up projections
  dim3 gI(II / 128, TOK / 128);
  k_gemm_bf16<true><<<gI, 256, 0, stream>>>(h2, wi0T, ub,  nullptr, 1.0f, TOK, II, HHD);
  k_gemm_bf16<true><<<gI, 256, 0, stream>>>(h2, wi1T, vgb, nullptr, 1.0f, TOK, II, HHD);

  // GeGLU + mid LN
  k_geglu_midln<<<TOK, 256, 0, stream>>>(ub, vgb, ln_mid, gb);

  // down projection + residual -> final f32 output
  k_gemm_bf16<false><<<gH, 256, 0, stream>>>(gb, woutT, (float*)d_out, x1, 1.0f, TOK, HHD, II);
}
